// HyenaBlock_29772713296393
// MI455X (gfx1250) — matrix operand order repaired and checked by reading, not compiled, not run on hardware
//
#include <hip/hip_runtime.h>
#include <hip/hip_bf16.h>
#include <math.h>

// ---------------------------------------------------------------------------
// Problem constants (HyenaBlock): B=4, L=4096, D=1024, ORDER=2, POS=65, HID=64
// ---------------------------------------------------------------------------
#define BB     4
#define LL     4096
#define DD     1024
#define ORDER  2
#define POSN   65
#define HID    64
#define H3     3072          // (ORDER+1)*D
#define OD     2048          // ORDER*D
#define BL     (BB*LL)       // 16384

typedef __attribute__((ext_vector_type(16))) _Float16 v16h;
typedef __attribute__((ext_vector_type(8)))  _Float16 v8h;
typedef __attribute__((ext_vector_type(8)))  float    v8f;

// ---------------------------------------------------------------------------
// WMMA fragment maps used here (gfx1250 wave32; contiguous k, not the ISA table order):
//  A (f16, 16Mx32K): K = j + (lane>=16 ? 16 : 0), M = lane&15
//  B (f16, 32Kx16N): K = j + (lane>=16 ? 16 : 0), N = lane&15
//  C/D (f32, 8 vgpr): vgpr r -> M = r + (lane>=16 ? 8:0), N = lane&15
// ---------------------------------------------------------------------------
__device__ __forceinline__ int kmapA(int lane, int j) {
    return j + (lane & 16);
}
__device__ __forceinline__ int kmapB(int lane, int j) {
    return j + (lane & 16);
}
__device__ __forceinline__ v8f wmma16(v16h a, v16h b, v8f c) {
    return __builtin_amdgcn_wmma_f32_16x16x32_f16(
        false, a, false, b, (short)0, c, false, false);
}
__device__ __forceinline__ float wsum32(float v) {
    #pragma unroll
    for (int o = 16; o > 0; o >>= 1) v += __shfl_xor(v, o, 32);
    return v;
}

// ---------------------------------------------------------------------------
// Weight pre-swizzle into fragment-native order: frag[tile][lane][16 halves].
// A v16h per lane is then a single coalesced 32B load in the hot loops.
// ---------------------------------------------------------------------------
// fc_w -> B fragments. tiles: kb (K/32 = 32) x nt (D/16 = 64)
__global__ void cvt_fcw_kernel(const float* __restrict__ fc_w, _Float16* __restrict__ fcwF) {
    int wave = threadIdx.x >> 5, lane = threadIdx.x & 31;
    int tile = blockIdx.x * 8 + wave;                 // 2048 tiles
    int kb = tile >> 6, nt = tile & 63;
    int n = nt * 16 + (lane & 15);
    _Float16* dst = fcwF + (size_t)tile * 512 + lane * 16;
    #pragma unroll
    for (int j = 0; j < 16; ++j)
        dst[j] = (_Float16)fc_w[(size_t)n * DD + kb * 32 + kmapB(lane, j)];
}
// fnn_wout -> B fragments. tiles: kb (64/32 = 2) x nt (2048/16 = 128)
__global__ void cvt_wout_kernel(const float* __restrict__ wout, _Float16* __restrict__ woutF) {
    int wave = threadIdx.x >> 5, lane = threadIdx.x & 31;
    int tile = blockIdx.x * 8 + wave;                 // 256 tiles
    int kb = tile >> 7, nt = tile & 127;
    int n = nt * 16 + (lane & 15);
    _Float16* dst = woutF + (size_t)tile * 512 + lane * 16;
    #pragma unroll
    for (int j = 0; j < 16; ++j)
        dst[j] = (_Float16)wout[(size_t)n * HID + kb * 32 + kmapB(lane, j)];
}
// h0 (linear [d][lag]) -> A fragments. tiles: dt (D/16 = 64) x lb (L/32 = 128)
__global__ void pack_h0_kernel(const _Float16* __restrict__ h0, _Float16* __restrict__ h0f) {
    int wave = threadIdx.x >> 5, lane = threadIdx.x & 31;
    int tile = blockIdx.x * 8 + wave;                 // 8192 tiles
    int dt = tile >> 7, lb = tile & 127;
    int d = dt * 16 + (lane & 15);
    int lag0 = lb * 32;
    _Float16* dst = h0f + (size_t)tile * 512 + lane * 16;
    #pragma unroll
    for (int j = 0; j < 16; ++j)
        dst[j] = h0[(size_t)d * LL + lag0 + kmapA(lane, j)];
}

// ---------------------------------------------------------------------------
// K1: filter generator.  Per 16-row t-tile: sine-MLP in LDS (VALU), then
//     WMMA (16 x 64) @ wout (64 x 2048), scale by decay, store h0/h1 (f16).
// ---------------------------------------------------------------------------
__global__ void filter_kernel(const float* __restrict__ pos_z, const float* __restrict__ pos_t,
                              const float* __restrict__ w0, const float* __restrict__ b0, const float* __restrict__ f0,
                              const float* __restrict__ w1, const float* __restrict__ b1, const float* __restrict__ f1,
                              const float* __restrict__ w2, const float* __restrict__ b2, const float* __restrict__ f2,
                              const _Float16* __restrict__ woutF, const float* __restrict__ alphas,
                              _Float16* __restrict__ h0out, _Float16* __restrict__ h1out) {
    __shared__ float zbuf[16][72];
    __shared__ float hA[16][68];
    __shared__ float hBuf[16][68];
    const int tid = threadIdx.x;
    const int t0  = blockIdx.x * 16;

    for (int i = tid; i < 16 * POSN; i += 256) {
        int r = i / POSN, p = i % POSN;
        zbuf[r][p] = pos_z[(t0 + r) * POSN + p];
    }
    __syncthreads();
    for (int i = tid; i < 16 * HID; i += 256) {       // layer 0: 65->64, sine
        int r = i >> 6, hh = i & 63;
        float acc = b0[hh];
        #pragma unroll 5
        for (int p = 0; p < POSN; ++p) acc += zbuf[r][p] * w0[hh * POSN + p];
        hA[r][hh] = __sinf(f0[hh] * acc);
    }
    __syncthreads();
    for (int i = tid; i < 16 * HID; i += 256) {       // layer 1
        int r = i >> 6, hh = i & 63;
        float acc = b1[hh];
        #pragma unroll 8
        for (int p = 0; p < HID; ++p) acc += hA[r][p] * w1[hh * HID + p];
        hBuf[r][hh] = __sinf(f1[hh] * acc);
    }
    __syncthreads();
    for (int i = tid; i < 16 * HID; i += 256) {       // layer 2
        int r = i >> 6, hh = i & 63;
        float acc = b2[hh];
        #pragma unroll 8
        for (int p = 0; p < HID; ++p) acc += hBuf[r][p] * w2[hh * HID + p];
        hA[r][hh] = __sinf(f2[hh] * acc);
    }
    __syncthreads();

    const int lane = tid & 31, wave = tid >> 5;
    const int m = lane & 15;
    v16h a_lo, a_hi;
    #pragma unroll
    for (int j = 0; j < 16; ++j) {
        int k = kmapA(lane, j);
        a_lo[j] = (_Float16)hA[m][k];
        a_hi[j] = (_Float16)hA[m][k + 32];
    }
    for (int nt = wave; nt < OD / 16; nt += 8) {
        // pre-swizzled B fragments: tiles (kb=0|1)*128 + nt
        v16h blo = *(const v16h*)(woutF + (size_t)nt * 512 + lane * 16);
        v16h bhi = *(const v16h*)(woutF + (size_t)(128 + nt) * 512 + lane * 16);
        v8f c = {};
        c = wmma16(a_lo, blo, c);
        c = wmma16(a_hi, bhi, c);
        const int n0 = nt * 16;
        #pragma unroll
        for (int r = 0; r < 8; ++r) {
            int t  = t0 + r + ((lane >> 4) << 3);
            int ch = n0 + (lane & 15);
            int io = ch >> 10, d = ch & (DD - 1);
            float val = c[r] * __expf(alphas[d] * pos_t[t]);   // decay
            _Float16* dst = (io == 0) ? h0out : h1out;
            dst[d * LL + t] = (_Float16)val;
        }
    }
}

// ---------------------------------------------------------------------------
// K2: LayerNorm(u) fused with the ONLY 3 projection channels used (0,1,3071).
// ---------------------------------------------------------------------------
__global__ void lnproj_kernel(const float* __restrict__ u,
                              const float* __restrict__ n1w, const float* __restrict__ n1b,
                              const float* __restrict__ proj_w, const float* __restrict__ proj_b,
                              float* __restrict__ xraw) {
    const int wave = threadIdx.x >> 5, lane = threadIdx.x & 31;
    const int row  = blockIdx.x * 8 + wave;
    const float* up = u + (size_t)row * DD;
    float vals[32];
    float s = 0.f;
    #pragma unroll
    for (int q = 0; q < 32; ++q) { vals[q] = up[q * 32 + lane]; s += vals[q]; }
    const float mean = wsum32(s) * (1.f / DD);
    float vv = 0.f;
    #pragma unroll
    for (int q = 0; q < 32; ++q) { float d = vals[q] - mean; vv += d * d; }
    const float rs = rsqrtf(wsum32(vv) * (1.f / DD) + 1e-5f);

    const float* pw0 = proj_w;
    const float* pw1 = proj_w + DD;
    const float* pwL = proj_w + (size_t)(H3 - 1) * DD;
    float a0 = 0.f, a1 = 0.f, aL = 0.f;
    #pragma unroll
    for (int q = 0; q < 32; ++q) {
        int d = q * 32 + lane;
        float xh = (vals[q] - mean) * rs * n1w[d] + n1b[d];
        a0 += xh * pw0[d]; a1 += xh * pw1[d]; aL += xh * pwL[d];
    }
    a0 = wsum32(a0); a1 = wsum32(a1); aL = wsum32(aL);
    if (lane == 0) {
        xraw[row]          = a0 + proj_b[0];
        xraw[BL + row]     = a1 + proj_b[1];
        xraw[2 * BL + row] = aL + proj_b[H3 - 1];
    }
}

// ---------------------------------------------------------------------------
// K3: causal 3-tap depthwise conv on the 3 live channels (+bias)
// ---------------------------------------------------------------------------
__global__ void dwconv_kernel(const float* __restrict__ xraw,
                              const float* __restrict__ conv_w, const float* __restrict__ conv_b,
                              float* __restrict__ aarr, _Float16* __restrict__ alast16) {
    const int idx = blockIdx.x * 256 + threadIdx.x;   // b*L + l
    const int l   = idx & (LL - 1);
    #pragma unroll
    for (int c3 = 0; c3 < 3; ++c3) {
        const int cch = (c3 == 2) ? (H3 - 1) : c3;
        const float* x = xraw + c3 * BL;
        float xm2 = (l >= 2) ? x[idx - 2] : 0.f;
        float xm1 = (l >= 1) ? x[idx - 1] : 0.f;
        float v = conv_w[cch * 3 + 0] * xm2 + conv_w[cch * 3 + 1] * xm1
                + conv_w[cch * 3 + 2] * x[idx] + conv_b[cch];
        aarr[c3 * BL + idx] = v;
        if (c3 == 2) alast16[idx] = (_Float16)v;
    }
}

// ---------------------------------------------------------------------------
// K4: causal conv #1 as WMMA GEMM per batch:
//     Y0[d,t] = sum_lag h0[d,lag]*x[t-lag]; A = pre-swizzled h0 fragments
//     (one v16h load/lane), B = Toeplitz(x) from LDS window.
//     Epilogue: v1[b,d,t] = a0[b,t]*(Y0 + alast[b,t]*hy_bias0[d]) (f16)
// ---------------------------------------------------------------------------
__global__ void conv1_kernel(const _Float16* __restrict__ h0f, const _Float16* __restrict__ alast16,
                             const float* __restrict__ a0arr, const float* __restrict__ hb0,
                             _Float16* __restrict__ v1) {
    __shared__ __align__(32) _Float16 xs[64 + LL];
    const int tid = threadIdx.x;
    const int b   = blockIdx.x >> 8;
    const int t0  = (blockIdx.x & 255) * 16;
    const int fill = 64 + t0 + 16;
    for (int i = tid; i < fill; i += 256)
        xs[i] = (i < 64) ? (_Float16)0.f : alast16[b * LL + (i - 64)];
    __syncthreads();

    const int lane = tid & 31, wave = tid >> 5;
    for (int dt = wave; dt < DD / 16; dt += 8) {
        const _Float16* afrag = h0f + ((size_t)dt * 128) * 512 + lane * 16;
        v8f acc = {};
        for (int lag0 = 0; lag0 <= t0 + 15; lag0 += 32) {
            __builtin_prefetch(afrag + 512, 0, 1);
            v16h a = *(const v16h*)afrag;             // coalesced 32B/lane
            afrag += 512;
            v16h bb;
            #pragma unroll
            for (int j = 0; j < 16; ++j) {            // reversed Toeplitz gather
                int idx = t0 + (lane & 15) - lag0 - kmapB(lane, j);   // >= -46
                bb[j] = xs[64 + idx];
            }
            acc = wmma16(a, bb, acc);
        }
        const int d0 = dt * 16;
        #pragma unroll
        for (int r = 0; r < 8; ++r) {
            int d = d0 + r + ((lane >> 4) << 3);
            int t = t0 + (lane & 15);
            float al = (float)xs[64 + t];
            float y0 = acc[r] + al * hb0[d];
            v1[(b * DD + d) * LL + t] = (_Float16)(a0arr[b * LL + t] * y0);
        }
    }
}

// ---------------------------------------------------------------------------
// K5: causal conv #2 (depthwise) as WMMA per d:
//     A = Toeplitz(h1_d) (16t x 32s) from LDS, B = v1[b,d,s-block] as one
//     v16h LDS read per lane (4 live of 16 N columns).
// ---------------------------------------------------------------------------
__global__ void conv2_kernel(const _Float16* __restrict__ h1, const _Float16* __restrict__ v1,
                             const float* __restrict__ a1arr, const float* __restrict__ hb1,
                             float* __restrict__ voutT) {
    __shared__ __align__(32) _Float16 hs[64 + LL + 16];
    __shared__ __align__(32) _Float16 vs[4][LL + 64];
    const int d   = blockIdx.x;
    const int tid = threadIdx.x;
    for (int i = tid; i < 64 + LL + 16; i += 256)
        hs[i] = (i < 64 || i >= 64 + LL) ? (_Float16)0.f : h1[d * LL + (i - 64)];
    for (int b = 0; b < BB; ++b)
        for (int i = tid; i < LL + 64; i += 256)
            vs[b][i] = (i < LL) ? v1[(b * DD + d) * LL + i] : (_Float16)0.f;
    __syncthreads();

    const int lane = tid & 31, wave = tid >> 5;
    const int m = lane & 15;
    const int n = lane & 15;
    for (int tt = wave; tt < LL / 16; tt += 8) {
        const int t0 = tt * 16;
        v8f acc = {};
        for (int s0 = 0; s0 <= t0 + 15; s0 += 32) {
            v16h a;
            #pragma unroll
            for (int j = 0; j < 16; ++j)              // reversed Toeplitz gather
                a[j] = hs[64 + t0 + m - s0 - kmapA(lane, j)];
            v16h bb = {};
            if (n < BB)
                bb = *(const v16h*)(&vs[n][s0 + (lane & 16)]);  // 32B LDS read
            acc = wmma16(a, bb, acc);
        }
        #pragma unroll
        for (int r = 0; r < 8; ++r) {
            int t = t0 + r + ((lane >> 4) << 3);
            int b = lane & 15;
            if (b < BB) {
                float vv = acc[r] + (float)vs[b][t] * hb1[d];
                voutT[(size_t)(b * LL + t) * DD + d] = a1arr[b * LL + t] * vv;
            }
        }
    }
}

// ---------------------------------------------------------------------------
// K6a: y = u + v^T, second LayerNorm; emit y (f32 residual) + yhat (f16 GEMM A)
// ---------------------------------------------------------------------------
__global__ void resid_ln_kernel(const float* __restrict__ u, const float* __restrict__ voutT,
                                const float* __restrict__ n2w, const float* __restrict__ n2b,
                                float* __restrict__ ybuf, _Float16* __restrict__ yhat) {
    const int wave = threadIdx.x >> 5, lane = threadIdx.x & 31;
    const int row  = blockIdx.x * 8 + wave;
    const float* up = u + (size_t)row * DD;
    const float* vp = voutT + (size_t)row * DD;
    float vals[32];
    float s = 0.f;
    #pragma unroll
    for (int q = 0; q < 32; ++q) {
        float y = up[q * 32 + lane] + vp[q * 32 + lane];
        vals[q] = y; s += y;
    }
    const float mean = wsum32(s) * (1.f / DD);
    float vv = 0.f;
    #pragma unroll
    for (int q = 0; q < 32; ++q) { float dd = vals[q] - mean; vv += dd * dd; }
    const float rs = rsqrtf(wsum32(vv) * (1.f / DD) + 1e-5f);
    float* yp = ybuf + (size_t)row * DD;
    _Float16* hp = yhat + (size_t)row * DD;
    #pragma unroll
    for (int q = 0; q < 32; ++q) {
        int d = q * 32 + lane;
        yp[d] = vals[q];
        hp[d] = (_Float16)((vals[q] - mean) * rs * n2w[d] + n2b[d]);
    }
}

// ---------------------------------------------------------------------------
// K6b: out = yhat @ fc_w^T + fc_b + y   (WMMA GEMM; A: two v8h LDS reads,
//      B: one v16h load from pre-swizzled fragments)
// ---------------------------------------------------------------------------
__global__ void fc_kernel(const _Float16* __restrict__ yhat, const _Float16* __restrict__ fcwF,
                          const float* __restrict__ fcb, const float* __restrict__ ybuf,
                          float* __restrict__ out) {
    __shared__ __align__(32) _Float16 As[16 * 1032];   // 1024 -> 1032 pad
    const int m0 = blockIdx.x * 16;
    for (int i = threadIdx.x; i < 16 * DD; i += 256) {
        int r = i >> 10, k = i & (DD - 1);
        As[r * 1032 + k] = yhat[(size_t)(m0 + r) * DD + k];
    }
    __syncthreads();

    const int lane = threadIdx.x & 31, wave = threadIdx.x >> 5;
    const int m  = lane & 15;
    const int hi = lane >> 4;
    for (int nt = wave; nt < DD / 16; nt += 8) {
        v8f acc = {};
        for (int k0 = 0; k0 < DD; k0 += 32) {
            const _Float16* ap = As + m * 1032 + k0;
            v8h alo = *(const v8h*)(ap + (hi ? 16 : 0));    // 16B LDS reads
            v8h ahi = *(const v8h*)(ap + (hi ? 24 : 8));
            v16h a;
            #pragma unroll
            for (int j = 0; j < 8; ++j) { a[j] = alo[j]; a[j + 8] = ahi[j]; }
            v16h bb = *(const v16h*)(fcwF + ((size_t)(k0 >> 5) * 64 + nt) * 512 + lane * 16);
            acc = wmma16(a, bb, acc);
        }
        const int n0 = nt * 16;
        #pragma unroll
        for (int r = 0; r < 8; ++r) {
            int row = m0 + r + (hi << 3);
            int col = n0 + (lane & 15);
            out[(size_t)row * DD + col] = acc[r] + fcb[col] + ybuf[(size_t)row * DD + col];
        }
    }
}

// ---------------------------------------------------------------------------
// Host launcher
// ---------------------------------------------------------------------------
extern "C" void kernel_launch(void* const* d_in, const int* in_sizes, int n_in,
                              void* d_out, int out_size, void* d_ws, size_t ws_size,
                              hipStream_t stream) {
    (void)in_sizes; (void)n_in; (void)out_size; (void)ws_size;
    const float* u       = (const float*)d_in[0];
    const float* n1w     = (const float*)d_in[1];
    const float* n1b     = (const float*)d_in[2];
    const float* proj_w  = (const float*)d_in[3];
    const float* proj_b  = (const float*)d_in[4];
    const float* conv_w  = (const float*)d_in[5];
    const float* conv_b  = (const float*)d_in[6];
    const float* pos_t   = (const float*)d_in[7];
    const float* pos_z   = (const float*)d_in[8];
    const float* fnn_w0  = (const float*)d_in[9];
    const float* fnn_b0  = (const float*)d_in[10];
    const float* freq0   = (const float*)d_in[11];
    const float* fnn_w1  = (const float*)d_in[12];
    const float* fnn_b1  = (const float*)d_in[13];
    const float* freq1   = (const float*)d_in[14];
    const float* fnn_w2  = (const float*)d_in[15];
    const float* fnn_b2  = (const float*)d_in[16];
    const float* freq2   = (const float*)d_in[17];
    const float* fnn_wout= (const float*)d_in[18];
    const float* alphas  = (const float*)d_in[19];
    const float* hy_bias = (const float*)d_in[20];
    const float* n2w     = (const float*)d_in[21];
    const float* n2b     = (const float*)d_in[22];
    const float* fc_w    = (const float*)d_in[23];
    const float* fc_b    = (const float*)d_in[24];
    float* out = (float*)d_out;

    // workspace carve-up (all chunk sizes are 256B multiples)
    char* p = (char*)d_ws;
    float*    xraw    = (float*)p;    p += (size_t)3 * BL * 4;
    float*    aarr    = (float*)p;    p += (size_t)3 * BL * 4;
    _Float16* alast16 = (_Float16*)p; p += (size_t)BL * 2;
    _Float16* h0      = (_Float16*)p; p += (size_t)DD * LL * 2;
    _Float16* h0f     = (_Float16*)p; p += (size_t)DD * LL * 2;   // fragments
    _Float16* h1      = (_Float16*)p; p += (size_t)DD * LL * 2;
    _Float16* v1      = (_Float16*)p; p += (size_t)BB * DD * LL * 2;
    float*    voutT   = (float*)p;    p += (size_t)BB * LL * DD * 4;
    float*    ybuf    = (float*)p;    p += (size_t)BB * LL * DD * 4;
    _Float16* yhat    = (_Float16*)p; p += (size_t)BB * LL * DD * 2;
    _Float16* fcwF    = (_Float16*)p; p += (size_t)DD * DD * 2;
    _Float16* woutF   = (_Float16*)p; p += (size_t)OD * HID * 2;

    // weight pre-swizzle into WMMA fragment layouts
    cvt_fcw_kernel <<<256, 256, 0, stream>>>(fc_w, fcwF);     // 2048 tiles
    cvt_wout_kernel<<<32, 256, 0, stream>>>(fnn_wout, woutF); // 256 tiles

    // filter MLP + decay -> h0/h1 (f16), then repack h0 into A-fragments
    filter_kernel<<<LL / 16, 256, 0, stream>>>(pos_z, pos_t,
        fnn_w0, fnn_b0, freq0, fnn_w1, fnn_b1, freq1, fnn_w2, fnn_b2, freq2,
        woutF, alphas, h0, h1);
    pack_h0_kernel<<<1024, 256, 0, stream>>>(h0, h0f);        // 8192 tiles

    // LN + 3 live projection channels -> xraw; 3-tap causal conv -> a0/a1/alast
    lnproj_kernel<<<BL / 8, 256, 0, stream>>>(u, n1w, n1b, proj_w, proj_b, xraw);
    dwconv_kernel<<<BL / 256, 256, 0, stream>>>(xraw, conv_w, conv_b, aarr, alast16);

    // causal long conv #1 (WMMA GEMM) -> v1 (f16)
    conv1_kernel<<<BB * (LL / 16), 256, 0, stream>>>(h0f, alast16, aarr /*a0*/, hy_bias, v1);

    // causal long conv #2 (depthwise WMMA) -> voutT (f32, transposed to (b,l,d))
    conv2_kernel<<<DD, 256, 0, stream>>>(h1, v1, aarr + BL /*a1*/, hy_bias + DD, voutT);

    // residual + LN2 -> ybuf/yhat; final FC GEMM + residual -> out
    resid_ln_kernel<<<BL / 8, 256, 0, stream>>>(u, voutT, n2w, n2b, ybuf, yhat);
    fc_kernel<<<BL / 16, 256, 0, stream>>>(yhat, fcwF, fc_b, ybuf, out);
}
